// StyleEncoder_44461501448527
// MI455X (gfx1250) — matrix operand order repaired and checked by reading, not compiled, not run on hardware
//
#include <hip/hip_runtime.h>
#include <hip/hip_bf16.h>

// ---------------------------------------------------------------------------
// StyleEncoder for MI455X (gfx1250, wave32, WMMA + TDM).
// GEMM-shaped math -> v_wmma_f32_16x16x32_f16 (f16 in, f32 acc).
// LDS staging of GEMM operands -> Tensor Data Mover (tensor_load_to_lds),
// with a plain vectorized-copy fallback when the builtin is unavailable.
// ---------------------------------------------------------------------------

typedef _Float16 half_t;
typedef __attribute__((ext_vector_type(16))) _Float16 v16h;
typedef __attribute__((ext_vector_type(8)))  _Float16 v8h;
typedef __attribute__((ext_vector_type(8)))  float    v8f;

typedef unsigned int u32x4 __attribute__((ext_vector_type(4)));
typedef int          i32x4 __attribute__((ext_vector_type(4)));
typedef int          i32x8 __attribute__((ext_vector_type(8)));

#if defined(__has_builtin)
#if __has_builtin(__builtin_amdgcn_tensor_load_to_lds) && __has_builtin(__builtin_amdgcn_s_wait_tensorcnt)
#define HAVE_TDM 1
#endif
#endif
#ifndef HAVE_TDM
#define HAVE_TDM 0
#endif

#define B_   32
#define C_   8
#define T_   8192
#define D_   64
#define TP_  512
#define TOK_ (B_*TP_*C_)     /* 131072 tokens for GAT-space GEMMs */
#define FFTN 4096

__device__ __forceinline__ float gelu_f(float x) {
    return 0.5f * x * (1.0f + erff(x * 0.70710678118654752440f));
}
__device__ __forceinline__ float sigm_f(float x) {
    return 1.0f / (1.0f + __expf(-x));
}

#if HAVE_TDM
// ---------------------------------------------------------------------------
// TDM: DMA `ndw` dwords of contiguous global memory into LDS.
// D# built per CDNA5 ISA: 2D tensor with a single row
// (tensor_dim0 = tile_dim0 = ndw dwords, tensor_dim1 = tile_dim1 = 1).
// Issue once per workgroup (wave 0); completion via s_wait_tensorcnt.
// ---------------------------------------------------------------------------
__device__ __forceinline__ void tdm_copy_dwords(unsigned lds_addr,
                                                unsigned long long gaddr,
                                                unsigned ndw)
{
    u32x4 g0 = {0u, 0u, 0u, 0u};
    g0[0] = 1u;                                                   // count=1 (valid user D#)
    g0[1] = lds_addr;                                             // D#.lds_addr (bytes)
    g0[2] = (unsigned)(gaddr & 0xffffffffull);                    // global_addr[31:0]
    g0[3] = (unsigned)((gaddr >> 32) & 0x1ffffffull) | (2u << 30);// global_addr[56:32] | type=2
    i32x8 g1 = {0, 0, 0, 0, 0, 0, 0, 0};
    g1[0] = (int)(2u << 16);                                      // data_size=2 -> 4-byte elems
    g1[1] = (int)((ndw & 0xffffu) << 16);                         // tensor_dim0[15:0]
    g1[2] = (int)(((ndw >> 16) & 0xffffu) | (1u << 16));          // tensor_dim0[31:16], tensor_dim1=1
    g1[3] = (int)((ndw & 0xffffu) << 16);                         // tile_dim0 = ndw
    g1[4] = 1;                                                    // tile_dim1 = 1
    g1[5] = (int)ndw;                                             // tensor_dim0_stride[31:0]
    i32x4 gz = {0, 0, 0, 0};
#if __clang_major__ >= 23
    i32x8 gz8 = {0, 0, 0, 0, 0, 0, 0, 0};
    __builtin_amdgcn_tensor_load_to_lds(g0, g1, gz, gz, gz8, 0);
#else
    __builtin_amdgcn_tensor_load_to_lds(g0, g1, gz, gz, 0);
#endif
}
#endif

// ---------------------------------------------------------------------------
// WMMA fragment loaders (CDNA5 16-bit layouts, CDNA5 ISA §7.12.2)
// A: 16x32 (MxK). lane<16: M=lane, K={k0..k0+7, k0+16..k0+23}
//                 lane>=16: M=lane-16, K={k0+8..k0+15, k0+24..k0+31}
// B: 32x16 (KxN) stored as weights W[N][K] (row-major N x K):
//                 lane<16: N=lane, K={k0..k0+7, k0+16..k0+23} ; lane>=16: N=lane-16, K={k0+8..k0+15, k0+24..k0+31}
// Both reduce to two contiguous 16-byte LDS loads per fragment.
// ---------------------------------------------------------------------------
__device__ __forceinline__ v16h load_a_frag(const half_t* A, int lda, int m0, int k0, int lane)
{
    int row = m0 + (lane & 15);
    int kb  = k0 + ((lane >> 4) << 3);
    const half_t* p = A + row * lda + kb;
    v8h lo = *(const v8h*)p;
    v8h hi = *(const v8h*)(p + 16);
    v16h r;
#pragma unroll
    for (int i = 0; i < 8; ++i) { r[i] = lo[i]; r[i + 8] = hi[i]; }
    return r;
}

__device__ __forceinline__ v16h load_b_frag(const half_t* Bw, int ldb, int n0, int k0, int lane)
{
    int n  = n0 + (lane & 15);
    int kb = k0 + ((lane >> 4) << 3);
    const half_t* p = Bw + n * ldb + kb;
    v8h lo = *(const v8h*)p;
    v8h hi = *(const v8h*)(p + 16);
    v16h r;
#pragma unroll
    for (int i = 0; i < 8; ++i) { r[i] = lo[i]; r[i + 8] = hi[i]; }
    return r;
}

// f32 -> f16 conversion (weights staged once)
__global__ void cvt16_kernel(const float* __restrict__ src, half_t* __restrict__ dst, int n)
{
    int i = blockIdx.x * 256 + threadIdx.x;
    if (i < n) dst[i] = (half_t)src[i];
}

// ---------------------------------------------------------------------------
// Stage 1: conv1d(1->32,k=7,p=3) + BN + GELU + maxpool4.  x:(256,8192) -> h1:(256,32,2048)
// ---------------------------------------------------------------------------
__global__ void conv1_pool_kernel(const float* __restrict__ x,
    const float* __restrict__ w1, const float* __restrict__ b1,
    const float* __restrict__ g1, const float* __restrict__ be1,
    const float* __restrict__ m1, const float* __restrict__ v1,
    float* __restrict__ h1)
{
    int row = blockIdx.y;               // 0..255 (b*8+c)
    int oc  = threadIdx.x & 31;
    int tl  = threadIdx.x >> 5;         // 0..7
    int tp  = blockIdx.x * 8 + tl;      // 0..2047
    const float* xr = x + (long long)row * T_;
    float w[7];
#pragma unroll
    for (int k = 0; k < 7; ++k) w[k] = w1[oc * 7 + k];
    float bias  = b1[oc];
    float scale = g1[oc] * rsqrtf(v1[oc] + 1e-5f);
    float shift = be1[oc] - m1[oc] * scale;
    float mx = -INFINITY;
#pragma unroll
    for (int q = 0; q < 4; ++q) {
        int t = tp * 4 + q;
        float s = bias;
#pragma unroll
        for (int k = 0; k < 7; ++k) {
            int tt = t + k - 3;
            float xv = (tt >= 0 && tt < T_) ? xr[tt] : 0.0f;
            s += xv * w[k];
        }
        mx = fmaxf(mx, gelu_f(s * scale + shift));
    }
    h1[((long long)row * 32 + oc) * 2048 + tp] = mx;
}

// ---------------------------------------------------------------------------
// Stage 2: conv1d(32->64,k=5,p=2) + BN + GELU + maxpool4. Writes (B,Tp,C,D) layout.
// ---------------------------------------------------------------------------
__global__ void conv2_pool_kernel(const float* __restrict__ h1,
    const float* __restrict__ w2, const float* __restrict__ b2,
    const float* __restrict__ g2, const float* __restrict__ be2,
    const float* __restrict__ m2, const float* __restrict__ v2,
    float* __restrict__ ht)
{
    __shared__ float sh[32][20];
    int row = blockIdx.y;               // 0..255
    int oc  = threadIdx.x & 63;
    int tl  = threadIdx.x >> 6;         // 0..3
    int tp  = blockIdx.x * 4 + tl;      // 0..511
    int t0  = blockIdx.x * 16 - 2;      // input t of sh[.][0]
    for (int i = threadIdx.x; i < 32 * 20; i += 256) {
        int ic = i / 20, lt = i % 20;
        int tt = t0 + lt;
        sh[ic][lt] = (tt >= 0 && tt < 2048) ? h1[((long long)row * 32 + ic) * 2048 + tt] : 0.0f;
    }
    __syncthreads();
    float acc0 = b2[oc], acc1 = b2[oc], acc2 = b2[oc], acc3 = b2[oc];
    for (int ic = 0; ic < 32; ++ic) {
#pragma unroll
        for (int k = 0; k < 5; ++k) {
            float w = w2[((long long)oc * 32 + ic) * 5 + k];
            int base = tl * 4 + k;
            acc0 += w * sh[ic][base + 0];
            acc1 += w * sh[ic][base + 1];
            acc2 += w * sh[ic][base + 2];
            acc3 += w * sh[ic][base + 3];
        }
    }
    float scale = g2[oc] * rsqrtf(v2[oc] + 1e-5f);
    float shift = be2[oc] - m2[oc] * scale;
    float mx = fmaxf(fmaxf(gelu_f(acc0 * scale + shift), gelu_f(acc1 * scale + shift)),
                     fmaxf(gelu_f(acc2 * scale + shift), gelu_f(acc3 * scale + shift)));
    int b = row >> 3, c = row & 7;
    ht[(((long long)b * TP_ + tp) * C_ + c) * D_ + oc] = mx;
}

// ---------------------------------------------------------------------------
// Feature-correlation adjacency term:  h_avg over Tp -> center -> normalize -> gram
// ---------------------------------------------------------------------------
__global__ void featcorr_kernel(const float* __restrict__ ht, float* __restrict__ featc)
{
    __shared__ float havg[C_][D_];
    __shared__ float nrm[C_];
    int b = blockIdx.x;
    int idx = threadIdx.x;              // 512 threads
    {
        int c = idx >> 6, d = idx & 63;
        float s = 0.f;
        for (int tp = 0; tp < TP_; ++tp)
            s += ht[(((long long)b * TP_ + tp) * C_ + c) * D_ + d];
        havg[c][d] = s * (1.0f / TP_);
    }
    __syncthreads();
    if (idx < C_) {
        int c = idx;
        float m = 0.f;
        for (int d = 0; d < D_; ++d) m += havg[c][d];
        m *= (1.0f / D_);
        float n2 = 0.f;
        for (int d = 0; d < D_; ++d) { float v = havg[c][d] - m; havg[c][d] = v; n2 += v * v; }
        nrm[c] = sqrtf(n2) + 1e-8f;
    }
    __syncthreads();
    if (idx < 64) {
        int c = idx >> 3, e = idx & 7;
        float s = 0.f;
        for (int d = 0; d < D_; ++d) s += havg[c][d] * havg[e][d];
        featc[b * 64 + idx] = s / (nrm[c] * nrm[e]);
    }
}

// ---------------------------------------------------------------------------
// |rfft(x, 8192)| via complex-packed 4096-pt radix-2 DIT FFT in LDS + unpack.
// ---------------------------------------------------------------------------
__global__ void rfft_mag_kernel(const float* __restrict__ x, float* __restrict__ spec)
{
    __shared__ float re[FFTN];
    __shared__ float im[FFTN];
    int row = blockIdx.x;
    int tid = threadIdx.x;
    const float* xr = x + (long long)row * T_;
    for (int n = tid; n < FFTN; n += 256) {
        int p = (int)(__brev((unsigned)n) >> 20);   // 12-bit reverse
        re[p] = xr[2 * n];
        im[p] = xr[2 * n + 1];
    }
    __syncthreads();
    for (int s = 1; s <= 12; ++s) {
        int half = 1 << (s - 1);
        for (int p = tid; p < FFTN / 2; p += 256) {
            int j  = p & (half - 1);
            int g  = p >> (s - 1);
            int i1 = (g << s) + j;
            int i2 = i1 + half;
            float ang = -3.14159265358979323846f * (float)j / (float)half;
            float si, co;
            __sincosf(ang, &si, &co);
            float tr = co * re[i2] - si * im[i2];
            float ti = co * im[i2] + si * re[i2];
            re[i2] = re[i1] - tr; im[i2] = im[i1] - ti;
            re[i1] += tr;         im[i1] += ti;
        }
        __syncthreads();
    }
    for (int k = tid; k <= FFTN; k += 256) {
        int kk = k & (FFTN - 1);
        int k2 = (FFTN - kk) & (FFTN - 1);
        float ar = re[kk], ai = im[kk];
        float br = re[k2], bi = -im[k2];
        float xer = 0.5f * (ar + br), xei = 0.5f * (ai + bi);
        float dr = ar - br, di = ai - bi;
        float xor_ = 0.5f * di, xoi = -0.5f * dr;
        float ang = -3.14159265358979323846f * (float)k / (float)FFTN;
        float si, co;
        __sincosf(ang, &si, &co);
        float tr = co * xor_ - si * xoi;
        float ti = co * xoi + si * xor_;
        float Xr = xer + tr, Xi = xei + ti;
        spec[(long long)row * 4097 + k] = sqrtf(Xr * Xr + Xi * Xi);
    }
}

// Spectral coherence gram (8x8 per batch over 4097 bins)
__global__ void speccoh_kernel(const float* __restrict__ spec, float* __restrict__ specc)
{
    __shared__ float dots[64];
    int b = blockIdx.x;
    int tid = threadIdx.x;              // 64 threads
    int c = tid >> 3, e = tid & 7;
    {
        const float* pc = spec + ((long long)b * 8 + c) * 4097;
        const float* pe = spec + ((long long)b * 8 + e) * 4097;
        float s = 0.f;
        for (int k = 0; k < 4097; ++k) s += pc[k] * pe[k];
        dots[tid] = s;
    }
    __syncthreads();
    float nc = sqrtf(dots[c * 8 + c]) + 1e-8f;
    float ne = sqrtf(dots[e * 8 + e]) + 1e-8f;
    specc[b * 64 + tid] = dots[tid] / (nc * ne);
}

// adj = softmax(log_w)[0]*featc + [1]*specc + [2]*(prior+prior^T); also to output
__global__ void adj_kernel(const float* __restrict__ featc, const float* __restrict__ specc,
                           const float* __restrict__ prior, const float* __restrict__ log_w,
                           float* __restrict__ adj, float* __restrict__ out_adj)
{
    int b = blockIdx.x;
    int tid = threadIdx.x;              // 64
    float l0 = log_w[0], l1 = log_w[1], l2 = log_w[2];
    float m  = fmaxf(l0, fmaxf(l1, l2));
    float e0 = __expf(l0 - m), e1 = __expf(l1 - m), e2 = __expf(l2 - m);
    float inv = 1.0f / (e0 + e1 + e2);
    int c = tid >> 3, e = tid & 7;
    float v = e0 * inv * featc[b * 64 + tid] + e1 * inv * specc[b * 64 + tid]
            + e2 * inv * (prior[c * 8 + e] + prior[e * 8 + c]);
    adj[b * 64 + tid]     = v;
    out_adj[b * 64 + tid] = v;
}

// ---------------------------------------------------------------------------
// LayerNorm over D=64, one token per thread; writes f16 (feeds WMMA GEMMs only).
// ---------------------------------------------------------------------------
__global__ void ln64_kernel(const float* __restrict__ X, const float* __restrict__ g,
                            const float* __restrict__ bta, half_t* __restrict__ Y)
{
    long long t = (long long)blockIdx.x * 256 + threadIdx.x;
    const float* xp = X + t * D_;
    float m = 0.f;
#pragma unroll 8
    for (int i = 0; i < D_; ++i) m += xp[i];
    m *= (1.0f / D_);
    float v = 0.f;
#pragma unroll 8
    for (int i = 0; i < D_; ++i) { float d = xp[i] - m; v += d * d; }
    float inv = rsqrtf(v * (1.0f / D_) + 1e-5f);
    half_t* yp = Y + t * D_;
#pragma unroll 8
    for (int i = 0; i < D_; ++i) yp[i] = (half_t)((xp[i] - m) * inv * g[i] + bta[i]);
}

// ---------------------------------------------------------------------------
// Generic WMMA token-GEMM, TDM-staged:
//   out[tokens,N] = act(X16[tokens,K] @ W16[N,K]^T + bias) (+resid)
// Block = 128 threads (4 waves), 128 tokens/block.  A (contiguous 128xK f16)
// and B (NxK f16 weights) are DMA'd into LDS by the Tensor Data Mover; wave 0
// issues both descriptors and waits on TENSORcnt, the rest sync at the barrier.
// Writes f32 Y or f16 Y16 (exactly one non-null).
// ---------------------------------------------------------------------------
__global__ void gemm_tok_kernel(const half_t* __restrict__ X16, const half_t* __restrict__ W16,
                                const float* __restrict__ bias, const float* __restrict__ resid,
                                float* __restrict__ Y, half_t* __restrict__ Y16,
                                int K, int N, int act)
{
    extern __shared__ __align__(16) char smem[];
    half_t* Ash = (half_t*)smem;         // [128][K]
    half_t* Bsh = Ash + 128 * K;         // [N][K]
    int tid = threadIdx.x;               // 128
    long long tb = (long long)blockIdx.x * 128;
#if HAVE_TDM
    if (tid < 32) {                      // wave 0 issues the DMA for the block
        tdm_copy_dwords((unsigned)(size_t)Ash,
                        (unsigned long long)(size_t)(X16 + tb * K),
                        (unsigned)(128 * K / 2));
        tdm_copy_dwords((unsigned)(size_t)Bsh,
                        (unsigned long long)(size_t)W16,
                        (unsigned)(N * K / 2));
        __builtin_amdgcn_s_wait_tensorcnt(0);
    }
#else
    for (int i = tid * 8; i < 128 * K; i += 128 * 8)
        *(v8h*)(Ash + i) = *(const v8h*)(X16 + tb * K + i);
    for (int i = tid * 8; i < N * K; i += 128 * 8)
        *(v8h*)(Bsh + i) = *(const v8h*)(W16 + i);
#endif
    __syncthreads();

    int wave = tid >> 5, lane = tid & 31;
    int ksteps = K >> 5, ntiles = N >> 4;
    for (int mi = 0; mi < 2; ++mi) {
        int m0 = (wave * 2 + mi) * 16;
        for (int nt = 0; nt < ntiles; ++nt) {
            v8f acc = {};
            for (int ks = 0; ks < ksteps; ++ks) {
                v16h a = load_a_frag(Ash, K, m0, ks * 32, lane);
                v16h b = load_b_frag(Bsh, K, nt * 16, ks * 32, lane);
                acc = __builtin_amdgcn_wmma_f32_16x16x32_f16(false, a, false, b,
                                                             (short)0, acc, false, false);
            }
            int nbase = nt * 16 + (lane & 15);
            int rbase = m0 + ((lane >> 4) << 3);
#pragma unroll
            for (int r = 0; r < 8; ++r) {
                long long trow = tb + rbase + r;
                float v = acc[r] + bias[nbase];
                if (act) v = gelu_f(v);
                long long oidx = trow * N + nbase;
                if (resid) v += resid[oidx];
                if (Y16) Y16[oidx] = (half_t)v;
                else     Y[oidx]   = v;
            }
        }
    }
}

// ---------------------------------------------------------------------------
// GAT attention over C=8 channels, 4 heads of dk=16, adj bias.  One thread per
// (b,t,c,head).  Writes f16 (feeds the out-projection WMMA GEMM).
// ---------------------------------------------------------------------------
__global__ void attn_kernel(const float* __restrict__ qkv, const float* __restrict__ adj,
                            half_t* __restrict__ attno)
{
    int idx = blockIdx.x * 256 + threadIdx.x;   // B*TP*C*4 = 524288
    int hh = idx & 3;
    int c  = (idx >> 2) & 7;
    int t  = (idx >> 5) & 511;
    int b  = idx >> 14;
    long long base = ((long long)b * TP_ + t) * C_;
    const float* q = qkv + (base + c) * 192 + hh * 16;
    float sc[8];
    float mx = -INFINITY;
    const float scale = 0.25f;                  // dk^-0.5, dk=16
    for (int e = 0; e < C_; ++e) {
        const float* kp = qkv + (base + e) * 192 + 64 + hh * 16;
        float s = 0.f;
#pragma unroll
        for (int d = 0; d < 16; ++d) s += q[d] * kp[d];
        s = s * scale + adj[b * 64 + c * 8 + e];
        sc[e] = s;
        mx = fmaxf(mx, s);
    }
    float den = 0.f;
    for (int e = 0; e < C_; ++e) { sc[e] = __expf(sc[e] - mx); den += sc[e]; }
    float inv = 1.0f / den;
    float out[16];
#pragma unroll
    for (int d = 0; d < 16; ++d) out[d] = 0.f;
    for (int e = 0; e < C_; ++e) {
        const float* vp = qkv + (base + e) * 192 + 128 + hh * 16;
        float a = sc[e] * inv;
#pragma unroll
        for (int d = 0; d < 16; ++d) out[d] += a * vp[d];
    }
    half_t* op = attno + (base + c) * 64 + hh * 16;
#pragma unroll
    for (int d = 0; d < 16; ++d) op[d] = (half_t)out[d];
}

// ---------------------------------------------------------------------------
// GRU scan.  gi = x@Wih^T + bih precomputed (parallel WMMA GEMM).  Each block
// owns 16 rows; Whh is TDM-staged to LDS once; per step one 16x192 WMMA GEMM
// (6 waves x 2 N-tiles) + fused gate math, barrier-synchronized.
// ---------------------------------------------------------------------------
__global__ void gru_scan_kernel(const float* __restrict__ gi, const half_t* __restrict__ whh16,
                                const float* __restrict__ bhh, float* __restrict__ hout, int dir)
{
    __shared__ __align__(16) half_t whhs[192 * 64];
    __shared__ __align__(16) half_t hsh[16 * 64];
    __shared__ float ghs[16 * 192];
    __shared__ float hcur[16 * 64];
    int tid = threadIdx.x;              // 192
    int rowbase = blockIdx.x * 16;
#if HAVE_TDM
    if (tid < 32) {
        tdm_copy_dwords((unsigned)(size_t)whhs,
                        (unsigned long long)(size_t)whh16,
                        (unsigned)(192 * 64 / 2));
        __builtin_amdgcn_s_wait_tensorcnt(0);
    }
#else
    for (int i = tid * 8; i < 192 * 64; i += 192 * 8)
        *(v8h*)(whhs + i) = *(const v8h*)(whh16 + i);
#endif
    for (int i = tid; i < 16 * 64; i += 192) { hcur[i] = 0.f; hsh[i] = (half_t)0.f; }
    __syncthreads();

    int wave = tid >> 5, lane = tid & 31;
    for (int ts = 0; ts < TP_; ++ts) {
        int t = dir ? (TP_ - 1 - ts) : ts;
        for (int nn = 0; nn < 2; ++nn) {
            int nt = wave * 2 + nn;
            v8f acc = {};
#pragma unroll
            for (int ks = 0; ks < 2; ++ks) {
                v16h a = load_a_frag(hsh, 64, 0, ks * 32, lane);
                v16h b = load_b_frag(whhs, 64, nt * 16, ks * 32, lane);
                acc = __builtin_amdgcn_wmma_f32_16x16x32_f16(false, a, false, b,
                                                             (short)0, acc, false, false);
            }
            int nbase = nt * 16 + (lane & 15);
            int rbase = (lane >> 4) << 3;
#pragma unroll
            for (int r = 0; r < 8; ++r)
                ghs[(rbase + r) * 192 + nbase] = acc[r];
        }
        __syncthreads();
        for (int i2 = tid; i2 < 16 * 64; i2 += 192) {
            int r = i2 >> 6, j = i2 & 63;
            int grow = rowbase + r;
            int b = grow >> 3, c = grow & 7;
            long long tok = ((long long)b * TP_ + t) * C_ + c;
            const float* gp = gi + tok * 192;
            // speculative prefetch of next timestep's gi row (dropped at edges)
            __builtin_prefetch(gp + (dir ? -(C_ * 192) : (C_ * 192)), 0, 1);
            float rr = sigm_f(gp[j]      + ghs[r * 192 + j]       + bhh[j]);
            float zz = sigm_f(gp[64 + j] + ghs[r * 192 + 64 + j]  + bhh[64 + j]);
            float nv = tanhf(gp[128 + j] + rr * (ghs[r * 192 + 128 + j] + bhh[128 + j]));
            float hn = (1.0f - zz) * nv + zz * hcur[i2];
            hcur[i2] = hn;
            hsh[i2]  = (half_t)hn;
        }
        __syncthreads();
    }
    for (int i2 = tid; i2 < 16 * 64; i2 += 192) {
        int r = i2 >> 6, j = i2 & 63;
        hout[(long long)(rowbase + r) * 128 + dir * 64 + j] = hcur[i2];
    }
}

// ---------------------------------------------------------------------------
// Gated readout. out: [readout 32x128][adj 32x64][gate 32x8]
// ---------------------------------------------------------------------------
__global__ void readout_kernel(const float* __restrict__ hout, const float* __restrict__ Wg,
                               const float* __restrict__ bg, float* __restrict__ out)
{
    __shared__ float gates[8];
    __shared__ float gsum;
    int b = blockIdx.x;
    int tid = threadIdx.x;              // 128
    if (tid < 8) {
        const float* hp = hout + (long long)(b * 8 + tid) * 128;
        float s = bg[0];
        for (int k = 0; k < 128; ++k) s += hp[k] * Wg[k];
        float g = sigm_f(s);
        gates[tid] = g;
        out[4096 + 2048 + b * 8 + tid] = g;
    }
    __syncthreads();
    if (tid == 0) {
        float s = 0.f;
        for (int c = 0; c < 8; ++c) s += gates[c];
        gsum = s + 1e-8f;
    }
    __syncthreads();
    float acc = 0.f;
    for (int c = 0; c < 8; ++c) acc += hout[(long long)(b * 8 + c) * 128 + tid] * gates[c];
    out[b * 128 + tid] = acc / gsum;
}

// ---------------------------------------------------------------------------
// Host launcher
// ---------------------------------------------------------------------------
extern "C" void kernel_launch(void* const* d_in, const int* in_sizes, int n_in,
                              void* d_out, int out_size, void* d_ws, size_t ws_size,
                              hipStream_t stream)
{
    (void)in_sizes; (void)n_in; (void)out_size; (void)ws_size;
    const float* x     = (const float*)d_in[0];
    const float* w1    = (const float*)d_in[1];
    const float* b1    = (const float*)d_in[2];
    const float* g1    = (const float*)d_in[3];
    const float* be1   = (const float*)d_in[4];
    const float* m1    = (const float*)d_in[5];
    const float* v1    = (const float*)d_in[6];
    const float* w2    = (const float*)d_in[7];
    const float* b2    = (const float*)d_in[8];
    const float* g2    = (const float*)d_in[9];
    const float* be2   = (const float*)d_in[10];
    const float* m2    = (const float*)d_in[11];
    const float* v2    = (const float*)d_in[12];
    const float* prior = (const float*)d_in[13];
    const float* log_w = (const float*)d_in[14];
    const float* gatn_g = (const float*)d_in[39];
    const float* gatn_b = (const float*)d_in[40];
    const float* wih_f  = (const float*)d_in[41];
    const float* whh_f  = (const float*)d_in[42];
    const float* bih_f  = (const float*)d_in[43];
    const float* bhh_f  = (const float*)d_in[44];
    const float* wih_b  = (const float*)d_in[45];
    const float* whh_b  = (const float*)d_in[46];
    const float* bih_b  = (const float*)d_in[47];
    const float* bhh_b  = (const float*)d_in[48];
    const float* Wgate  = (const float*)d_in[49];
    const float* bgate  = (const float*)d_in[50];

    char*  wsb = (char*)d_ws;
    float* out = (float*)d_out;

    // ---- workspace layout (bytes, 256-aligned) ----
    const size_t SZB_H1   = 256ull * 32 * 2048 * 4;     // 67,108,864
    const size_t SZB_HT   = (size_t)TOK_ * D_ * 4;      // 33,554,432
    const size_t SZB_SPEC = 256ull * 4097 * 4;          //  4,195,328
    const size_t OFFB_H1    = 0;
    const size_t OFFB_HT    = OFFB_H1 + SZB_H1;
    const size_t OFFB_SPEC  = OFFB_HT + SZB_HT;
    const size_t OFFB_FEATC = OFFB_SPEC + SZB_SPEC;
    const size_t OFFB_SPECC = OFFB_FEATC + 8192;
    const size_t OFFB_ADJ   = OFFB_SPECC + 8192;
    const size_t OFFB_QKV   = OFFB_ADJ + 8192;
    const size_t OFFB_GIB   = OFFB_QKV + (size_t)TOK_ * 192 * 4;
    const size_t OFFB_HOUT  = OFFB_GIB + (size_t)TOK_ * 192 * 4;
    const size_t OFFB_HN16  = OFFB_HOUT + 256ull * 128 * 4;
    const size_t OFFB_W16   = OFFB_HN16 + (size_t)TOK_ * D_ * 2;
    const size_t OFFB_ATTNO16 = OFFB_H1;                        // reuse (h1 dead)
    const size_t OFFB_FFN116  = OFFB_H1 + (size_t)TOK_ * D_ * 2;
    const size_t OFFB_GIF     = OFFB_QKV;                       // reuse (qkv dead)

    float*  h1      = (float*)(wsb + OFFB_H1);
    float*  ht      = (float*)(wsb + OFFB_HT);
    float*  spec    = (float*)(wsb + OFFB_SPEC);
    float*  featc   = (float*)(wsb + OFFB_FEATC);
    float*  specc   = (float*)(wsb + OFFB_SPECC);
    float*  adj     = (float*)(wsb + OFFB_ADJ);
    float*  qkv     = (float*)(wsb + OFFB_QKV);
    float*  gif     = (float*)(wsb + OFFB_GIF);
    float*  gib     = (float*)(wsb + OFFB_GIB);
    float*  hout    = (float*)(wsb + OFFB_HOUT);
    half_t* hn16    = (half_t*)(wsb + OFFB_HN16);
    half_t* attno16 = (half_t*)(wsb + OFFB_ATTNO16);
    half_t* ffn116  = (half_t*)(wsb + OFFB_FFN116);
    half_t* w16     = (half_t*)(wsb + OFFB_W16);

    // f16 weight pool offsets (elements): per layer Wqkv,Wout,W1,W2; then GRU mats
    const size_t WQKV[2] = {0, 32768};
    const size_t WOUT[2] = {12288, 12288 + 32768};
    const size_t WW1[2]  = {16384, 16384 + 32768};
    const size_t WW2[2]  = {24576, 24576 + 32768};
    const size_t WIHF = 65536, WIHB = 77824, WHHF = 90112, WHHB = 102400;

    // ---- CNN front-end ----
    conv1_pool_kernel<<<dim3(256, 256), 256, 0, stream>>>(x, w1, b1, g1, be1, m1, v1, h1);
    conv2_pool_kernel<<<dim3(128, 256), 256, 0, stream>>>(h1, w2, b2, g2, be2, m2, v2, ht);

    // ---- adjacency ----
    featcorr_kernel<<<32, 512, 0, stream>>>(ht, featc);
    rfft_mag_kernel<<<256, 256, 0, stream>>>(x, spec);
    speccoh_kernel<<<32, 64, 0, stream>>>(spec, specc);
    adj_kernel<<<32, 64, 0, stream>>>(featc, specc, prior, log_w, adj, out + 4096);

    // ---- stage f16 weights once ----
    for (int l = 0; l < 2; ++l) {
        int base = 15 + 12 * l;
        cvt16_kernel<<<48, 256, 0, stream>>>((const float*)d_in[base + 2],  w16 + WQKV[l], 192 * 64);
        cvt16_kernel<<<16, 256, 0, stream>>>((const float*)d_in[base + 4],  w16 + WOUT[l],  64 * 64);
        cvt16_kernel<<<32, 256, 0, stream>>>((const float*)d_in[base + 8],  w16 + WW1[l],  128 * 64);
        cvt16_kernel<<<32, 256, 0, stream>>>((const float*)d_in[base + 10], w16 + WW2[l],  64 * 128);
    }
    cvt16_kernel<<<48, 256, 0, stream>>>(wih_f, w16 + WIHF, 192 * 64);
    cvt16_kernel<<<48, 256, 0, stream>>>(wih_b, w16 + WIHB, 192 * 64);
    cvt16_kernel<<<48, 256, 0, stream>>>(whh_f, w16 + WHHF, 192 * 64);
    cvt16_kernel<<<48, 256, 0, stream>>>(whh_b, w16 + WHHB, 192 * 64);

    // ---- 2 GAT layers (TDM-staged WMMA GEMMs) ----
    const int GRID_T = TOK_ / 128;   // 1024 token blocks
    for (int l = 0; l < 2; ++l) {
        int base = 15 + 12 * l;
        const float* ln1_g = (const float*)d_in[base + 0];
        const float* ln1_b = (const float*)d_in[base + 1];
        const float* bqkv  = (const float*)d_in[base + 3];
        const float* bout  = (const float*)d_in[base + 5];
        const float* ln2_g = (const float*)d_in[base + 6];
        const float* ln2_b = (const float*)d_in[base + 7];
        const float* b1p   = (const float*)d_in[base + 9];
        const float* b2p   = (const float*)d_in[base + 11];

        ln64_kernel<<<TOK_ / 256, 256, 0, stream>>>(ht, ln1_g, ln1_b, hn16);
        gemm_tok_kernel<<<GRID_T, 128, (128 * 64 + 192 * 64) * sizeof(half_t), stream>>>(
            hn16, w16 + WQKV[l], bqkv, nullptr, qkv, nullptr, 64, 192, 0);
        attn_kernel<<<TOK_ * 4 / 256, 256, 0, stream>>>(qkv, adj, attno16);
        gemm_tok_kernel<<<GRID_T, 128, (128 * 64 + 64 * 64) * sizeof(half_t), stream>>>(
            attno16, w16 + WOUT[l], bout, ht, ht, nullptr, 64, 64, 0);
        ln64_kernel<<<TOK_ / 256, 256, 0, stream>>>(ht, ln2_g, ln2_b, hn16);
        gemm_tok_kernel<<<GRID_T, 128, (128 * 64 + 128 * 64) * sizeof(half_t), stream>>>(
            hn16, w16 + WW1[l], b1p, nullptr, nullptr, ffn116, 64, 128, 1);
        gemm_tok_kernel<<<GRID_T, 128, (128 * 128 + 64 * 128) * sizeof(half_t), stream>>>(
            ffn116, w16 + WW2[l], b2p, ht, ht, nullptr, 128, 64, 0);
    }

    // ---- final LN + GRU ----
    ln64_kernel<<<TOK_ / 256, 256, 0, stream>>>(ht, gatn_g, gatn_b, hn16);
    gemm_tok_kernel<<<GRID_T, 128, (128 * 64 + 192 * 64) * sizeof(half_t), stream>>>(
        hn16, w16 + WIHF, bih_f, nullptr, gif, nullptr, 64, 192, 0);
    gemm_tok_kernel<<<GRID_T, 128, (128 * 64 + 192 * 64) * sizeof(half_t), stream>>>(
        hn16, w16 + WIHB, bih_b, nullptr, gib, nullptr, 64, 192, 0);
    gru_scan_kernel<<<16, 192, 0, stream>>>(gif, w16 + WHHF, bhh_f, hout, 0);
    gru_scan_kernel<<<16, 192, 0, stream>>>(gib, w16 + WHHB, bhh_b, hout, 1);

    // ---- gated readout ----
    readout_kernel<<<32, 128, 0, stream>>>(hout, Wgate, bgate, out);
}
